// Encoder_74783970558006
// MI455X (gfx1250) — matrix operand order repaired and checked by reading, not compiled, not run on hardware
//
#include <hip/hip_runtime.h>

typedef __attribute__((ext_vector_type(16))) __bf16 v16bf;
typedef __attribute__((ext_vector_type(8)))  __bf16 v8bf;
typedef __attribute__((ext_vector_type(4)))  __bf16 v4bf;
typedef __attribute__((ext_vector_type(8)))  float  v8f;
typedef __attribute__((ext_vector_type(4)))  int    i4_;

#define N_   2048
#define D_   256
#define H_   1024
#define OPT_ 64
#define S_   8

// ---------------- CDNA5 async global->LDS copy helpers ----------------
#if __has_builtin(__builtin_amdgcn_global_load_async_to_lds_b128)
#define HAVE_ASYNC_LDS 1
#endif

__device__ inline void cp16_g2l(const void* g, void* l) {
#if defined(HAVE_ASYNC_LDS)
  // signature (from hipcc diagnostic): (AS1 int4*, AS3 int4*, imm offset, imm cpol)
  __builtin_amdgcn_global_load_async_to_lds_b128(
      (__attribute__((address_space(1))) i4_*)g,
      (__attribute__((address_space(3))) i4_*)l, 0, 0);
#else
  *(v8bf*)l = *(const v8bf*)g;
#endif
}

__device__ inline void cp_wait() {
#if defined(HAVE_ASYNC_LDS)
#if __has_builtin(__builtin_amdgcn_s_wait_asynccnt)
  __builtin_amdgcn_s_wait_asynccnt(0);
#else
  asm volatile("s_wait_asynccnt 0x0" ::: "memory");
#endif
#endif
}

// ---------------- WMMA fragment loads (ISA 7.12.2 layouts) ----------------
union Frag { v16bf v; v8bf h[2]; };

// A 16x32 bf16: lane row M=lane&15; elems 0-7 -> K=(lane>>4)*8.. ; 8-15 -> 16+(lane>>4)*8..
__device__ inline v16bf load_a_frag(const __bf16* Ab, int mt, int lane) {
  int half = lane >> 4, row = mt * 16 + (lane & 15);
  Frag f;
  f.h[0] = *(const v8bf*)(Ab + row * 32 + half * 8);
  f.h[1] = *(const v8bf*)(Ab + row * 32 + 16 + half * 8);
  return f.v;
}
// B 32x16 bf16: lane col N=lane&15; elems 0-7 -> K=(lane>>4)*8.. ; 8-15 -> 16+(lane>>4)*8.. Bt stored [col][32k].
__device__ inline v16bf load_b_frag(const __bf16* Bt, int colbase, int lane) {
  int half = lane >> 4, col = colbase + (lane & 15);
  Frag f;
  f.h[0] = *(const v8bf*)(Bt + col * 32 + half * 8);
  f.h[1] = *(const v8bf*)(Bt + col * 32 + 16 + half * 8);
  return f.v;
}

__global__ void k_zero(float* p, int n) {
  int i = blockIdx.x * blockDim.x + threadIdx.x;
  if (i < n) p[i] = 0.f;
}

// ---------------- weight pre-conversion: f32 [K][NC] -> bf16 [K/32][NC][32] ----------------
__global__ __launch_bounds__(256)
void k_conv(const float* __restrict__ src, __bf16* __restrict__ dst, int K, int NC) {
  int id = blockIdx.x * 256 + threadIdx.x;
  int col = id % NC;
  int kb  = id / NC;
  if (kb * 32 >= K) return;
  __attribute__((aligned(16))) __bf16 tmp[32];
#pragma unroll
  for (int kk = 0; kk < 32; ++kk)
    tmp[kk] = (__bf16)src[(size_t)(kb * 32 + kk) * NC + col];
  v8bf* d = (v8bf*)(dst + ((size_t)kb * NC + col) * 32);
  d[0] = *(v8bf*)&tmp[0];  d[1] = *(v8bf*)&tmp[8];
  d[2] = *(v8bf*)&tmp[16]; d[3] = *(v8bf*)&tmp[24];
}

// ============ PRE-CONVERTED-WEIGHT PATH ============

// base_out = relu(cur @ base_w + bb), B from bf16 tiled layout. Tile 32x128, 8 waves.
__global__ __launch_bounds__(256)
void k_base_pre(const float* __restrict__ cur, const __bf16* __restrict__ wb,
                const float* __restrict__ bb, __bf16* __restrict__ bout) {
  __shared__ __bf16 Ab[32 * 32];
  __shared__ __bf16 Bt[128 * 32];
  int t = threadIdx.x, lane = t & 31, w = t >> 5;
  int row0 = blockIdx.x * 32, col0 = blockIdx.y * 128;
  int mt = w & 1, ng = w >> 1;
  v8f acc0 = {}, acc1 = {};
  for (int k0 = 0; k0 < 256; k0 += 32) {
    __syncthreads();
    { // A: 32x32 f32 -> bf16 (changes per stage; converted in-kernel)
      int r = t >> 3, kc = (t & 7) * 4;
      const float4 a = *(const float4*)(cur + (size_t)(row0 + r) * 256 + k0 + kc);
      v4bf p = { (__bf16)a.x, (__bf16)a.y, (__bf16)a.z, (__bf16)a.w };
      *(v4bf*)(Ab + r * 32 + kc) = p;
    }
    { // B: straight copy of pre-tiled bf16 (async global->LDS)
      int c = t >> 1, hh = (t & 1) * 16;
      const __bf16* src = wb + ((size_t)(k0 >> 5) * 1024 + col0 + c) * 32 + hh;
      cp16_g2l(src,     Bt + c * 32 + hh);
      cp16_g2l(src + 8, Bt + c * 32 + hh + 8);
    }
    cp_wait();
    __syncthreads();
    v16bf a  = load_a_frag(Ab, mt, lane);
    v16bf b0 = load_b_frag(Bt, ng * 32, lane);
    v16bf b1 = load_b_frag(Bt, ng * 32 + 16, lane);
    acc0 = __builtin_amdgcn_wmma_f32_16x16x32_bf16(false, a, false, b0, (short)0, acc0, false, false);
    acc1 = __builtin_amdgcn_wmma_f32_16x16x32_bf16(false, a, false, b1, (short)0, acc1, false, false);
  }
  int half = lane >> 4, nlo = lane & 15;
  int m = row0 + mt * 16 + half * 8;
  int c0 = col0 + ng * 32 + nlo;
  float bb0 = bb[c0], bb1 = bb[c0 + 16];
  for (int r = 0; r < 8; ++r) {
    float v0 = acc0[r] + bb0; v0 = v0 > 0.f ? v0 : 0.f;
    float v1 = acc1[r] + bb1; v1 = v1 > 0.f ? v1 : 0.f;
    bout[(size_t)(m + r) * 1024 + c0]      = (__bf16)v0;
    bout[(size_t)(m + r) * 1024 + c0 + 16] = (__bf16)v1;
  }
}

// Fused GEMM2 + per-option MSE loss, B pre-converted. WG = 32 rows x 8 options.
__global__ __launch_bounds__(256)
void k_loss_pre(const __bf16* __restrict__ bout, const __bf16* __restrict__ wl,
                const float* __restrict__ lb, const float* __restrict__ sb,
                const float* __restrict__ cur, const float* __restrict__ inp,
                float* __restrict__ losses, int s) {
  __shared__ __bf16 Ab[32 * 32];
  __shared__ __bf16 Bt[256 * 32];
  __shared__ float lpart[32];
  int t = threadIdx.x, lane = t & 31, w = t >> 5;
  int row0 = blockIdx.x * 32;
  int og0 = blockIdx.y * 8;
  int mt = w & 1, ng = w >> 1;           // wave = 16 rows x 64 cols (4 N-tiles)
  int half = lane >> 4, nlo = lane & 15;
  float ci[4][8];                        // (cur - inp): option-invariant
  for (int j = 0; j < 4; ++j) {
    int d = ng * 64 + j * 16 + nlo;
    for (int r = 0; r < 8; ++r) {
      int m = row0 + mt * 16 + half * 8 + r;
      ci[j][r] = cur[(size_t)m * 256 + d] - inp[(size_t)m * 256 + d];
    }
  }
  v8f vzero = {};
  for (int oi = 0; oi < 8; ++oi) {
    int opt = og0 + oi;
    v8f acc[4] = { vzero, vzero, vzero, vzero };
    __syncthreads();
    if (t < 32) lpart[t] = 0.f;
    for (int k0 = 0; k0 < 1024; k0 += 32) {
      __syncthreads();
      if (t < 128) { // A: 2KB bf16 copy (async)
        int r = t >> 2, kc = (t & 3) * 8;
        cp16_g2l(bout + (size_t)(row0 + r) * 1024 + k0 + kc, Ab + r * 32 + kc);
      }
      { // B: 16KB pre-tiled bf16 copy, thread t owns one column's 64B (async)
        const __bf16* src = wl + ((size_t)(k0 >> 5) * 16384 + opt * 256 + t) * 32;
        __bf16* d = Bt + t * 32;
        cp16_g2l(src,      d);
        cp16_g2l(src + 8,  d + 8);
        cp16_g2l(src + 16, d + 16);
        cp16_g2l(src + 24, d + 24);
      }
      cp_wait();
      __syncthreads();
      v16bf a = load_a_frag(Ab, mt, lane);
      for (int j = 0; j < 4; ++j) {
        v16bf b = load_b_frag(Bt, ng * 64 + j * 16, lane);
        acc[j] = __builtin_amdgcn_wmma_f32_16x16x32_bf16(false, a, false, b, (short)0, acc[j], false, false);
      }
    }
    float lbsb[4];
    for (int j = 0; j < 4; ++j) {
      int d = ng * 64 + j * 16 + nlo;
      lbsb[j] = lb[opt * 256 + d] + sb[opt * 256 + d];
    }
    for (int r = 0; r < 8; ++r) {
      float p = 0.f;
      for (int j = 0; j < 4; ++j) {
        float e = acc[j][r] + lbsb[j] + ci[j][r];
        p += e * e;
      }
      p += __shfl_xor(p, 1);
      p += __shfl_xor(p, 2);
      p += __shfl_xor(p, 4);
      p += __shfl_xor(p, 8);
      if (nlo == 0) atomicAdd(&lpart[mt * 16 + half * 8 + r], p);  // ds_add_f32
    }
    __syncthreads();
    if (t < 32) {
      int n = row0 + t;
      losses[(size_t)n * S_ * OPT_ + (size_t)s * OPT_ + opt] = lpart[t] * (1.0f / 256.0f);
    }
  }
}

// ============ FALLBACK PATH (in-kernel f32->bf16 conversion) ============

__global__ __launch_bounds__(256)
void k_base(const float* __restrict__ cur, const float* __restrict__ bw,
            const float* __restrict__ bb, __bf16* __restrict__ bout) {
  __shared__ __bf16 Ab[32 * 32];
  __shared__ __bf16 Bt[128 * 32];
  int t = threadIdx.x, lane = t & 31, w = t >> 5;
  int row0 = blockIdx.x * 32, col0 = blockIdx.y * 128;
  int mt = w & 1, ng = w >> 1;
  v8f acc0 = {}, acc1 = {};
  for (int k0 = 0; k0 < 256; k0 += 32) {
    __syncthreads();
    { int r = t >> 3, kc = (t & 7) * 4;
      const float4 a = *(const float4*)(cur + (size_t)(row0 + r) * 256 + k0 + kc);
      v4bf p = { (__bf16)a.x, (__bf16)a.y, (__bf16)a.z, (__bf16)a.w };
      *(v4bf*)(Ab + r * 32 + kc) = p; }
    { int c4 = (t & 31) * 4, rb = t >> 5;
      for (int q = 0; q < 4; ++q) {
        int r = rb + 8 * q;
        const float4 b = *(const float4*)(bw + (size_t)(k0 + r) * 1024 + col0 + c4);
        Bt[(c4 + 0) * 32 + r] = (__bf16)b.x;
        Bt[(c4 + 1) * 32 + r] = (__bf16)b.y;
        Bt[(c4 + 2) * 32 + r] = (__bf16)b.z;
        Bt[(c4 + 3) * 32 + r] = (__bf16)b.w;
      } }
    __syncthreads();
    v16bf a  = load_a_frag(Ab, mt, lane);
    v16bf b0 = load_b_frag(Bt, ng * 32, lane);
    v16bf b1 = load_b_frag(Bt, ng * 32 + 16, lane);
    acc0 = __builtin_amdgcn_wmma_f32_16x16x32_bf16(false, a, false, b0, (short)0, acc0, false, false);
    acc1 = __builtin_amdgcn_wmma_f32_16x16x32_bf16(false, a, false, b1, (short)0, acc1, false, false);
  }
  int half = lane >> 4, nlo = lane & 15;
  int m = row0 + mt * 16 + half * 8;
  int c0 = col0 + ng * 32 + nlo;
  float bb0 = bb[c0], bb1 = bb[c0 + 16];
  for (int r = 0; r < 8; ++r) {
    float v0 = acc0[r] + bb0; v0 = v0 > 0.f ? v0 : 0.f;
    float v1 = acc1[r] + bb1; v1 = v1 > 0.f ? v1 : 0.f;
    bout[(size_t)(m + r) * 1024 + c0]      = (__bf16)v0;
    bout[(size_t)(m + r) * 1024 + c0 + 16] = (__bf16)v1;
  }
}

__global__ __launch_bounds__(256)
void k_loss(const __bf16* __restrict__ bout, const float* __restrict__ lw,
            const float* __restrict__ lb, const float* __restrict__ sb,
            const float* __restrict__ cur, const float* __restrict__ inp,
            float* __restrict__ losses, int s) {
  __shared__ __bf16 Ab[32 * 32];
  __shared__ __bf16 Bt[256 * 32];
  __shared__ float lpart[32];
  int t = threadIdx.x, lane = t & 31, w = t >> 5;
  int row0 = blockIdx.x * 32;
  int og0 = blockIdx.y * 8;
  int mt = w & 1, ng = w >> 1;
  int half = lane >> 4, nlo = lane & 15;
  float ci[4][8];
  for (int j = 0; j < 4; ++j) {
    int d = ng * 64 + j * 16 + nlo;
    for (int r = 0; r < 8; ++r) {
      int m = row0 + mt * 16 + half * 8 + r;
      ci[j][r] = cur[(size_t)m * 256 + d] - inp[(size_t)m * 256 + d];
    }
  }
  v8f vzero = {};
  for (int oi = 0; oi < 8; ++oi) {
    int opt = og0 + oi;
    v8f acc[4] = { vzero, vzero, vzero, vzero };
    __syncthreads();
    if (t < 32) lpart[t] = 0.f;
    for (int k0 = 0; k0 < 1024; k0 += 32) {
      __syncthreads();
      { int r = t >> 3, kc = (t & 7) * 4;
        *(v4bf*)(Ab + r * 32 + kc) =
            *(const v4bf*)(bout + (size_t)(row0 + r) * 1024 + k0 + kc); }
      { int c4 = (t & 63) * 4, rb = t >> 6;
        for (int q = 0; q < 8; ++q) {
          int r = rb + 4 * q;
          const float4 b = *(const float4*)(lw + (size_t)(k0 + r) * 16384 + opt * 256 + c4);
          Bt[(c4 + 0) * 32 + r] = (__bf16)b.x;
          Bt[(c4 + 1) * 32 + r] = (__bf16)b.y;
          Bt[(c4 + 2) * 32 + r] = (__bf16)b.z;
          Bt[(c4 + 3) * 32 + r] = (__bf16)b.w;
        } }
      __syncthreads();
      v16bf a = load_a_frag(Ab, mt, lane);
      for (int j = 0; j < 4; ++j) {
        v16bf b = load_b_frag(Bt, ng * 64 + j * 16, lane);
        acc[j] = __builtin_amdgcn_wmma_f32_16x16x32_bf16(false, a, false, b, (short)0, acc[j], false, false);
      }
    }
    float lbsb[4];
    for (int j = 0; j < 4; ++j) {
      int d = ng * 64 + j * 16 + nlo;
      lbsb[j] = lb[opt * 256 + d] + sb[opt * 256 + d];
    }
    for (int r = 0; r < 8; ++r) {
      float p = 0.f;
      for (int j = 0; j < 4; ++j) {
        float e = acc[j][r] + lbsb[j] + ci[j][r];
        p += e * e;
      }
      p += __shfl_xor(p, 1);
      p += __shfl_xor(p, 2);
      p += __shfl_xor(p, 4);
      p += __shfl_xor(p, 8);
      if (nlo == 0) atomicAdd(&lpart[mt * 16 + half * 8 + r], p);
    }
    __syncthreads();
    if (t < 32) {
      int n = row0 + t;
      losses[(size_t)n * S_ * OPT_ + (size_t)s * OPT_ + opt] = lpart[t] * (1.0f / 256.0f);
    }
  }
}

// Per-row argmin + winning-option GEMV recompute.
__global__ __launch_bounds__(256)
void k_select(const __bf16* __restrict__ bout, const float* __restrict__ lw,
              const float* __restrict__ lb, const float* __restrict__ sb,
              float* __restrict__ cur, const float* __restrict__ losses,
              int* __restrict__ enc, float* __restrict__ recon, int s, int last) {
  __shared__ float sl[OPT_];
  __shared__ int bopt;
  __shared__ __bf16 sA[256];
  int t = threadIdx.x;
  int n = blockIdx.x;
  if (t < OPT_) sl[t] = losses[(size_t)n * S_ * OPT_ + (size_t)s * OPT_ + t];
  __syncthreads();
  if (t == 0) {
    float best = sl[0]; int bi = 0;
    for (int o = 1; o < OPT_; ++o) if (sl[o] < best) { best = sl[o]; bi = o; }
    bopt = bi;
    enc[n * S_ + s] = bi;
  }
  __syncthreads();
  int opt = bopt;
  float acc = 0.f;
  int d = t;
  for (int h0 = 0; h0 < H_; h0 += 256) {
    __syncthreads();
    sA[t] = bout[(size_t)n * H_ + h0 + t];
    __syncthreads();
    for (int i = 0; i < 256; ++i) {
      acc += (float)sA[i] * lw[(size_t)(h0 + i) * 16384 + opt * 256 + d];
    }
  }
  float out = cur[(size_t)n * 256 + d] + sb[opt * 256 + d] + lb[opt * 256 + d] + acc;
  cur[(size_t)n * 256 + d] = out;
  if (last) recon[(size_t)n * 256 + d] = out;
}

extern "C" void kernel_launch(void* const* d_in, const int* in_sizes, int n_in,
                              void* d_out, int out_size, void* d_ws, size_t ws_size,
                              hipStream_t stream) {
  (void)in_sizes; (void)n_in; (void)out_size;
  const float* inputs     = (const float*)d_in[0];
  const float* base_w     = (const float*)d_in[1];
  const float* base_b     = (const float*)d_in[2];
  const float* layer_w    = (const float*)d_in[3];
  const float* layer_b    = (const float*)d_in[4];
  const float* stage_bias = (const float*)d_in[5];

  // workspace layout
  size_t off_cur  = 0;
  size_t off_bout = off_cur  + (size_t)N_ * D_ * sizeof(float);           // 2 MB
  size_t off_wb   = off_bout + (size_t)N_ * H_ * sizeof(__bf16);          // +4 MB
  size_t off_wl   = off_wb   + (size_t)D_ * H_ * sizeof(__bf16);          // +0.5 MB
  size_t need     = off_wl   + (size_t)H_ * OPT_ * D_ * sizeof(__bf16);   // +33.5 MB
  bool pre = ws_size >= need;

  float*  cur  = (float*)((char*)d_ws + off_cur);
  __bf16* bout = (__bf16*)((char*)d_ws + off_bout);
  __bf16* wb   = (__bf16*)((char*)d_ws + off_wb);
  __bf16* wl   = (__bf16*)((char*)d_ws + off_wl);

  // d_out: [encodings N*S int] [recon N*D f32] [all_losses N*S*OPT f32]
  int*   enc    = (int*)d_out;
  float* recon  = (float*)d_out + (size_t)N_ * S_;
  float* losses = (float*)d_out + (size_t)N_ * S_ + (size_t)N_ * D_;

  k_zero<<<(N_ * D_ + 255) / 256, 256, 0, stream>>>(cur, N_ * D_);
  if (pre)
    k_conv<<<(8 * 1024) / 256, 256, 0, stream>>>(base_w, wb, 256, 1024);
  for (int s = 0; s < S_; ++s) {
    const float* lwf = layer_w    + (size_t)s * H_ * OPT_ * D_;
    const float* lb  = layer_b    + (size_t)s * OPT_ * D_;
    const float* sb  = stage_bias + (size_t)s * OPT_ * D_;
    if (pre) {
      k_conv<<<(32 * 16384) / 256, 256, 0, stream>>>(lwf, wl, 1024, 16384);
      k_base_pre<<<dim3(N_ / 32, H_ / 128), 256, 0, stream>>>(cur, wb, base_b, bout);
      k_loss_pre<<<dim3(N_ / 32, OPT_ / 8), 256, 0, stream>>>(bout, wl, lb, sb, cur, inputs, losses, s);
    } else {
      k_base<<<dim3(N_ / 32, H_ / 128), 256, 0, stream>>>(cur, base_w, base_b, bout);
      k_loss<<<dim3(N_ / 32, OPT_ / 8), 256, 0, stream>>>(bout, lwf, lb, sb, cur, inputs, losses, s);
    }
    k_select<<<N_, 256, 0, stream>>>(bout, lwf, lb, sb, cur, losses, enc, recon, s, s == S_ - 1);
  }
}
